// SMLP_MemQuant_37005438222904
// MI455X (gfx1250) — compile-verified, matrix operand order checked by reading, not run on hardware
//
#include <hip/hip_runtime.h>
#include <hip/hip_bf16.h>
#include <stdint.h>

// SNN forward for MI455X (gfx1250, wave32).
// Layer-1/2 GEMMs via V_WMMA_F32_16X16X32_F16; LIF membrane state lives in the
// WMMA accumulator registers across the whole 20-step scan; xi/spikes staged
// in LDS; W1/W2 pre-converted to padded f16 in workspace.
//
// Block = 160 threads = 5 wave32; each wave owns exactly 5 of the 25 hidden
// N-tiles (compile-time constant -> full unroll, no dynamic VGPR indexing).

typedef __attribute__((ext_vector_type(16))) _Float16 v16h;
typedef __attribute__((ext_vector_type(8)))  float    v8f;

#define B_TOT   16384
#define IN_DIM  784
#define KPAD1   800     // 25 k-tiles of 32 (cols 784..799 zero)
#define HID     400     // 25 n-tiles of 16 (exact)
#define KPAD2   416     // 13 k-tiles of 32 (cols 400..415 zero)
#define OUTD    10
#define THRESH  0.5f
#define DECAY   0.2f

#define NWAVES  5
#define TPW     5                       // n-tiles per wave (5*5 = 25)
#define NTHREADS (NWAVES * 32)

__device__ __forceinline__ float rng01(uint32_t step, uint32_t idx) {
  uint32_t h = idx * 0x9E3779B1u + (step + 1u) * 0x85EBCA77u + 0x165667B1u;
  h ^= h >> 16; h *= 0x85EBCA6Bu;
  h ^= h >> 13; h *= 0xC2B2AE35u;
  h ^= h >> 16;
  return (float)(h >> 8) * (1.0f / 16777216.0f);
}

// K offset inside a 32-wide k-tile for VGPR v and lane-half (ISA 16-bit A/B
// layout): lanes 0-15: V0..V3 -> K=0..7 pairs, V4..V7 -> K=16..23; +8 for hi.
__device__ __forceinline__ int kmap(int v, int half) {
  return ((v < 4) ? (2 * v) : (16 + 2 * (v - 4))) + 8 * half;
}

// Assemble a 16-element f16 fragment via 8x 32-bit loads of f16 pairs.
// The two 4-dword runs coalesce into two b128 loads.
__device__ __forceinline__ v16h load_frag_h(const _Float16* base, int stride_h,
                                            int row, int k0, int half) {
  union { v16h h; uint32_t u[8]; } f;
#pragma unroll
  for (int v = 0; v < 8; ++v) {
    int k = k0 + kmap(v, half);
    f.u[v] = *reinterpret_cast<const uint32_t*>(base + (size_t)row * stride_h + k);
  }
  return f.h;
}

__global__ __launch_bounds__(256) void cvt_w1_f16(const float* __restrict__ w,
                                                  _Float16* __restrict__ o) {
  int t = blockIdx.x * 256 + threadIdx.x;
  if (t >= HID * KPAD1) return;
  int r = t / KPAD1, c = t - r * KPAD1;
  o[t] = (c < IN_DIM) ? (_Float16)w[(size_t)r * IN_DIM + c] : (_Float16)0.0f;
}

__global__ __launch_bounds__(256) void cvt_w2_f16(const float* __restrict__ w,
                                                  _Float16* __restrict__ o) {
  int t = blockIdx.x * 256 + threadIdx.x;
  if (t >= 16 * KPAD2) return;
  int r = t / KPAD2, c = t - r * KPAD2;
  o[t] = (r < OUTD && c < HID) ? (_Float16)w[(size_t)r * HID + c] : (_Float16)0.0f;
}

__global__ __launch_bounds__(NTHREADS) void snn_scan_kernel(
    const float* __restrict__ x, const _Float16* __restrict__ w1h,
    const float* __restrict__ b1, const _Float16* __restrict__ w2h,
    const float* __restrict__ b2, const int* __restrict__ twp,
    float* __restrict__ out) {
  __shared__ _Float16 xi_s[16 * KPAD1];   // 25.6 KB: input spikes, f16
  __shared__ _Float16 sp_s[16 * KPAD2];   // 13.3 KB: layer-1 spikes, f16

  const int tid  = threadIdx.x;
  const int wave = tid >> 5;              // 0..4, wave-uniform
  const int lane = tid & 31;
  const int half = (lane >> 4) & 1;
  const int n16  = lane & 15;
  const int brow = blockIdx.x * 16;       // this block's 16 batch rows
  const int T    = twp[0];

  // Zero the K-pad columns once (never rewritten).
  for (int t = tid; t < 16 * (KPAD1 - IN_DIM); t += NTHREADS) {
    int r = t / (KPAD1 - IN_DIM), c = t - r * (KPAD1 - IN_DIM);
    xi_s[r * KPAD1 + IN_DIM + c] = (_Float16)0.0f;
  }
  for (int t = tid; t < 16 * (KPAD2 - HID); t += NTHREADS) {
    int r = t / (KPAD2 - HID), c = t - r * (KPAD2 - HID);
    sp_s[r * KPAD2 + HID + c] = (_Float16)0.0f;
  }

  // Per-wave bias scalars: C-matrix columns map to lanes, so b1 is one scalar
  // per lane per tile.
  float b1sc[TPW];
#pragma unroll
  for (int u = 0; u < TPW; ++u)
    b1sc[u] = b1[(wave * TPW + u) * 16 + n16];

  // Layer-1 membrane state, resident in WMMA accumulator layout:
  // entry d of a lane = element (M = d + 8*half, N = n16) of tile (wave*TPW+u).
  v8f mem[TPW] = {};

  // Layer-2 state (only wave 0 uses it).
  v8f h2m = {}, h2sum = {};
  const float b2sc = (n16 < OUTD) ? b2[n16] : 0.0f;

  for (int step = 0; step < T; ++step) {
    // ---- Bernoulli spike-encode x into LDS (cooperative) ----
    for (int t = tid; t < 16 * IN_DIM; t += NTHREADS) {
      int r = t / IN_DIM, c = t - r * IN_DIM;
      uint32_t gidx = (uint32_t)((brow + r) * IN_DIM + c);
      float xv = x[(size_t)(brow + r) * IN_DIM + c];
      float uv = rng01((uint32_t)step, gidx);
      xi_s[r * KPAD1 + c] = (_Float16)((xv > uv) ? 1.0f : 0.0f);
    }
    __syncthreads();

    // ---- Layer 1: mem <- decay(mem)*(1-spike) + b1, accumulate xi @ W1^T ----
#pragma unroll
    for (int u = 0; u < TPW; ++u) {
#pragma unroll
      for (int i = 0; i < 8; ++i) {
        float m = mem[u][i];
        mem[u][i] = ((m > THRESH) ? 0.0f : m * DECAY) + b1sc[u];
      }
    }
    for (int kt = 0; kt < 25; ++kt) {
      v16h a = load_frag_h(xi_s, KPAD1, n16, kt * 32, half);    // A from LDS
      int ktn = (kt + 1 < 25) ? (kt + 1) : kt;                  // clamp, no branch
      __builtin_prefetch(w1h + (size_t)(wave * TPW * 16 + n16) * KPAD1 + ktn * 32,
                         0, 1);
#pragma unroll
      for (int u = 0; u < TPW; ++u) {
        int wrow = (wave * TPW + u) * 16 + n16;                 // B row = hid unit
        v16h b = load_frag_h(w1h, KPAD1, wrow, kt * 32, half);
        mem[u] = __builtin_amdgcn_wmma_f32_16x16x32_f16(
            false, a, false, b, (short)0, mem[u], false, false);
      }
    }
    // Publish spikes (pure function of mem) to LDS for layer 2.
#pragma unroll
    for (int u = 0; u < TPW; ++u) {
#pragma unroll
      for (int d = 0; d < 8; ++d) {
        _Float16 s = (mem[u][d] > THRESH) ? (_Float16)1.0f : (_Float16)0.0f;
        sp_s[(d + 8 * half) * KPAD2 + (wave * TPW + u) * 16 + n16] = s;
      }
    }
    __syncthreads();

    // ---- Layer 2 on wave 0 (wave-uniform branch; EXEC stays all-1s) ----
    if (wave == 0) {
      v8f c2;
#pragma unroll
      for (int i = 0; i < 8; ++i) {
        float m = h2m[i];
        c2[i] = ((m > THRESH) ? 0.0f : m * DECAY) + b2sc;
      }
#pragma unroll
      for (int kt = 0; kt < 13; ++kt) {
        v16h a = load_frag_h(sp_s, KPAD2, n16, kt * 32, half);  // spikes (LDS)
        v16h b = load_frag_h(w2h, KPAD2, n16, kt * 32, half);   // W2 row (out)
        c2 = __builtin_amdgcn_wmma_f32_16x16x32_f16(
            false, a, false, b, (short)0, c2, false, false);
      }
      h2m = c2;
#pragma unroll
      for (int i = 0; i < 8; ++i)
        h2sum[i] += (c2[i] > THRESH) ? 1.0f : 0.0f;
    }
    // sp_s reads by wave 0 finish before anyone rewrites sp_s: the rewrite
    // happens after the post-xi __syncthreads of the next step, which wave 0
    // only reaches after completing layer 2.
  }

  // ---- Emit rate-coded output ----
  if (wave == 0) {
    float inv = 1.0f / (float)T;
#pragma unroll
    for (int d = 0; d < 8; ++d) {
      int m = d + 8 * half;
      if (n16 < OUTD)
        out[(size_t)(brow + m) * OUTD + n16] = h2sum[d] * inv;
    }
  }
}

extern "C" void kernel_launch(void* const* d_in, const int* in_sizes, int n_in,
                              void* d_out, int out_size, void* d_ws, size_t ws_size,
                              hipStream_t stream) {
  const float* x  = (const float*)d_in[0];   // (16384, 784)
  const float* W1 = (const float*)d_in[1];   // (400, 784)
  const float* b1 = (const float*)d_in[2];   // (400,)
  const float* W2 = (const float*)d_in[3];   // (10, 400)
  const float* b2 = (const float*)d_in[4];   // (10,)
  const int*   tw = (const int*)d_in[5];     // time_window (scalar)
  float* out = (float*)d_out;                // (16384, 10)

  // Workspace: padded f16 weights (W1: 400x800, W2: 16x416) ~ 653 KB.
  _Float16* w1h = (_Float16*)d_ws;
  _Float16* w2h = (_Float16*)((char*)d_ws + (size_t)HID * KPAD1 * sizeof(_Float16));

  hipLaunchKernelGGL(cvt_w1_f16, dim3((HID * KPAD1 + 255) / 256), dim3(256), 0,
                     stream, W1, w1h);
  hipLaunchKernelGGL(cvt_w2_f16, dim3((16 * KPAD2 + 255) / 256), dim3(256), 0,
                     stream, W2, w2h);
  hipLaunchKernelGGL(snn_scan_kernel, dim3(B_TOT / 16), dim3(NTHREADS), 0, stream,
                     x, w1h, b1, w2h, b2, tw, out);
}
